// DiseaseGNN_28578712387808
// MI455X (gfx1250) — matrix operand order repaired and checked by reading, not compiled, not run on hardware
//
#include <hip/hip_runtime.h>
#include <hip/hip_bf16.h>

// ---------------------------------------------------------------------------
// 2-layer GCN forward for MI455X (gfx1250, wave32).
//   h   = relu( Agg( x@W1 ) + b1 )      Agg = sym-normalized scatter-sum+self
//   out = Agg( h@W2 ) + b2
// GEMM1: v_wmma_f32_16x16x32_bf16, W1 staged in LDS (bf16, transposed),
// fully unrolled K loop. Scatters: native global_atomic_add_f32 (inline asm),
// accumulators are L2-resident (12.8MB << 192MB L2).
// ---------------------------------------------------------------------------

typedef __attribute__((ext_vector_type(16))) __bf16          v16bf;
typedef __attribute__((ext_vector_type(8)))  float           v8f;
typedef __attribute__((ext_vector_type(16))) unsigned short  v16us;

__device__ __forceinline__ unsigned short f32_to_bf16_rne(float f) {
  union { float f; unsigned u; } cv; cv.f = f;
  unsigned u = cv.u;
  u += 0x7FFFu + ((u >> 16) & 1u);          // round-to-nearest-even
  return (unsigned short)(u >> 16);
}

__device__ __forceinline__ void atomAddF(float* p, float v) {
  // no-return HW float atomic (STOREcnt path); avoids any CAS-loop lowering
  asm volatile("global_atomic_add_f32 %0, %1, off" :: "v"(p), "v"(v) : "memory");
}

// ---- degree / normalization ------------------------------------------------
__global__ void k_deg_init(float* __restrict__ deg, int n) {
  int i = blockIdx.x * blockDim.x + threadIdx.x;
  if (i < n) deg[i] = 1.0f;                 // self loop
}

__global__ void k_deg_count(const int* __restrict__ dst, float* __restrict__ deg, int e) {
  int i = blockIdx.x * blockDim.x + threadIdx.x;
  if (i < e) atomAddF(&deg[dst[i]], 1.0f);
}

__global__ void k_deg_rsqrt(float* __restrict__ deg, int n) {
  int i = blockIdx.x * blockDim.x + threadIdx.x;
  if (i < n) deg[i] = rsqrtf(deg[i]);       // in-place: deg -> dinv
}

// ---- GEMM1: h_pre = x @ W1 (WMMA bf16), fused self-loop init of agg1 -------
// One wave -> 16 rows x 32 cols (two 16x16 WMMA tiles), K fully unrolled 4x32.
__global__ __launch_bounds__(256) void k_gemm1_wmma(
    const float* __restrict__ x,     // [N,128] row-major
    const float* __restrict__ W1,    // [128,32] row-major (K x N)
    const float* __restrict__ dinv,  // [N]
    float* __restrict__ hpre,        // [N,32]
    float* __restrict__ agg1,        // [N,32]  init: hpre * dinv^2 (self loop)
    int n) {
  // Stage W1 in LDS as bf16, transposed: Wt[col][K], col<32, K<128 (8KB).
  __shared__ __bf16 Wt[32 * 128];
  for (int idx = threadIdx.x; idx < 32 * 128; idx += 256) {
    int col = idx >> 7;                     // idx / 128
    int K   = idx & 127;
    unsigned short u = f32_to_bf16_rne(W1[K * 32 + col]);
    Wt[idx] = __builtin_bit_cast(__bf16, u);
  }
  __syncthreads();

  const int lane    = threadIdx.x & 31;
  const int wave    = threadIdx.x >> 5;
  const int rowbase = (blockIdx.x * 8 + wave) * 16;
  if (rowbase >= n) return;

  const int hsel  = lane >> 4;      // 0: lanes 0-15, 1: lanes 16-31
  const int l16   = lane & 15;
  const int akoff = hsel * 16;      // A frag: upper half-wave holds K+16..31
  const int bkoff = hsel * 16;      // B frag: upper half-wave holds K+16..31

  int arow = rowbase + l16;
  if (arow >= n) arow = n - 1;      // safe duplicate (tiles are full: N%16==0)
  const float* __restrict__ xrow = x + (size_t)arow * 128;

  v8f c0 = {0.f, 0.f, 0.f, 0.f, 0.f, 0.f, 0.f, 0.f};
  v8f c1 = c0;

#pragma unroll
  for (int kk = 0; kk < 128; kk += 32) {
    __builtin_prefetch(xrow + kk + 32, 0, 0);     // global_prefetch_b8

    // A fragment: one contiguous 16-float run -> 4x global_load_b128 + cvt.
    // e=0..15 -> K=kk+akoff+e
    const float4 a0 = *(const float4*)(xrow + kk + akoff);
    const float4 a1 = *(const float4*)(xrow + kk + akoff + 4);
    const float4 a2 = *(const float4*)(xrow + kk + akoff + 8);
    const float4 a3 = *(const float4*)(xrow + kk + akoff + 12);
    const float af[16] = {a0.x, a0.y, a0.z, a0.w, a1.x, a1.y, a1.z, a1.w,
                          a2.x, a2.y, a2.z, a2.w, a3.x, a3.y, a3.z, a3.w};
    v16us au;
#pragma unroll
    for (int j = 0; j < 16; ++j) au[j] = f32_to_bf16_rne(af[j]);
    v16bf a = __builtin_bit_cast(v16bf, au);

    // B fragments from LDS: col = l16 (+16), K = kk+bkoff+j, j=0..15
    // contiguous 32B per fragment -> 2x ds_load_b128 each
    v16bf b0 = *(const v16bf*)(Wt + l16 * 128 + kk + bkoff);
    v16bf b1 = *(const v16bf*)(Wt + (l16 + 16) * 128 + kk + bkoff);

    c0 = __builtin_amdgcn_wmma_f32_16x16x32_bf16(false, a, false, b0, (short)0, c0, false, false);
    c1 = __builtin_amdgcn_wmma_f32_16x16x32_bf16(false, a, false, b1, (short)0, c1, false, false);
  }

  // C/D layout: VGPR r, lane -> M = rowbase + r + hsel*8, Ncol = l16 (+16 for c1)
  if (rowbase + 16 <= n) {          // full tile: straight-line stores
#pragma unroll
    for (int r = 0; r < 8; ++r) {
      int row = rowbase + r + hsel * 8;
      float dv = dinv[row];
      float sc = dv * dv;
      size_t base = (size_t)row * 32;
      hpre[base + l16]      = c0[r];
      hpre[base + l16 + 16] = c1[r];
      agg1[base + l16]      = c0[r] * sc;
      agg1[base + l16 + 16] = c1[r] * sc;
    }
  } else {
#pragma unroll
    for (int r = 0; r < 8; ++r) {
      int row = rowbase + r + hsel * 8;
      if (row < n) {
        float dv = dinv[row];
        float sc = dv * dv;
        size_t base = (size_t)row * 32;
        hpre[base + l16]      = c0[r];
        hpre[base + l16 + 16] = c1[r];
        agg1[base + l16]      = c0[r] * sc;
        agg1[base + l16 + 16] = c1[r] * sc;
      }
    }
  }
}

// ---- layer-1 scatter: one lane per (edge, channel) -------------------------
__global__ __launch_bounds__(256) void k_scatter1(
    const int* __restrict__ src, const int* __restrict__ dst,
    const float* __restrict__ hpre, const float* __restrict__ dinv,
    float* __restrict__ agg1, int e) {
  long long idx = (long long)blockIdx.x * blockDim.x + threadIdx.x;
  int ei = (int)(idx >> 5);
  int c  = (int)(idx & 31);
  if (ei >= e) return;
  int s = src[ei], d = dst[ei];
  float w = dinv[s] * dinv[d];
  atomAddF(&agg1[(size_t)d * 32 + c], hpre[(size_t)s * 32 + c] * w);
}

// ---- bias + relu + layer-2 projection (wave per node) ----------------------
__global__ __launch_bounds__(256) void k_relu_proj(
    const float* __restrict__ agg1, const float* __restrict__ b1,
    const float* __restrict__ W2,   const float* __restrict__ dinv,
    float* __restrict__ hout,       // d_out[0 : N*32)
    float* __restrict__ spre,       // [N] = h @ W2
    float* __restrict__ agg2,       // [N] init: spre * dinv^2 (self loop)
    int n) {
  int node = (int)(((long long)blockIdx.x * blockDim.x + threadIdx.x) >> 5);
  int lane = threadIdx.x & 31;
  if (node >= n) return;
  float v = agg1[(size_t)node * 32 + lane] + b1[lane];
  v = fmaxf(v, 0.0f);
  hout[(size_t)node * 32 + lane] = v;
  float p = v * W2[lane];
#pragma unroll
  for (int off = 16; off > 0; off >>= 1) p += __shfl_down(p, off, 32);
  if (lane == 0) {
    float dv = dinv[node];
    spre[node] = p;
    agg2[node] = p * dv * dv;
  }
}

// ---- layer-2 scatter: one thread per edge ----------------------------------
__global__ __launch_bounds__(256) void k_scatter2(
    const int* __restrict__ src, const int* __restrict__ dst,
    const float* __restrict__ spre, const float* __restrict__ dinv,
    float* __restrict__ agg2, int e) {
  int i = blockIdx.x * blockDim.x + threadIdx.x;
  if (i >= e) return;
  int s = src[i], d = dst[i];
  atomAddF(&agg2[d], spre[s] * dinv[s] * dinv[d]);
}

__global__ void k_final(const float* __restrict__ agg2, const float* __restrict__ b2,
                        float* __restrict__ scores, int n) {
  int i = blockIdx.x * blockDim.x + threadIdx.x;
  if (i < n) scores[i] = agg2[i] + b2[0];
}

// ---------------------------------------------------------------------------
extern "C" void kernel_launch(void* const* d_in, const int* in_sizes, int n_in,
                              void* d_out, int out_size, void* d_ws, size_t ws_size,
                              hipStream_t stream) {
  const float* x  = (const float*)d_in[0];   // [N,128]
  const int*   ei = (const int*)d_in[1];     // [2,E]
  const float* W1 = (const float*)d_in[2];   // [128,32]
  const float* b1 = (const float*)d_in[3];   // [32]
  const float* W2 = (const float*)d_in[4];   // [32]
  const float* b2 = (const float*)d_in[5];   // [1]

  const int N = in_sizes[0] / 128;
  const int E = in_sizes[1] / 2;
  const int* src = ei;
  const int* dst = ei + E;

  // workspace layout (floats): dinv[N] | hpre[N*32] | agg1[N*32] | spre[N] | agg2[N]
  float* ws   = (float*)d_ws;
  float* dinv = ws;
  float* hpre = dinv + (size_t)N;
  float* agg1 = hpre + (size_t)N * 32;
  float* spre = agg1 + (size_t)N * 32;
  float* agg2 = spre + (size_t)N;

  float* hout = (float*)d_out;               // [N,32]
  float* sout = hout + (size_t)N * 32;       // [N]

  const int T = 256;
  // 1) degrees -> dinv
  k_deg_init<<<(N + T - 1) / T, T, 0, stream>>>(dinv, N);
  k_deg_count<<<(E + T - 1) / T, T, 0, stream>>>(dst, dinv, E);
  k_deg_rsqrt<<<(N + T - 1) / T, T, 0, stream>>>(dinv, N);

  // 2) h_pre = x@W1 (WMMA) + fused self-loop init of agg1
  int tiles16 = (N + 15) / 16;
  k_gemm1_wmma<<<(tiles16 + 7) / 8, T, 0, stream>>>(x, W1, dinv, hpre, agg1, N);

  // 3) edge scatter, layer 1 (lane per channel)
  long long work1 = (long long)E * 32;
  k_scatter1<<<(int)((work1 + T - 1) / T), T, 0, stream>>>(src, dst, hpre, dinv, agg1, E);

  // 4) bias + relu + h@W2 projection + self-loop init of agg2
  k_relu_proj<<<(N + 7) / 8, T, 0, stream>>>(agg1, b1, W2, dinv, hout, spre, agg2, N);

  // 5) edge scatter, layer 2 (scalar)
  k_scatter2<<<(E + T - 1) / T, T, 0, stream>>>(src, dst, spre, dinv, agg2, E);

  // 6) bias -> scores
  k_final<<<(N + T - 1) / T, T, 0, stream>>>(agg2, b2, sout, N);
}
